// Attention_12987981103776
// MI455X (gfx1250) — matrix operand order repaired and checked by reading, not compiled, not run on hardware
//
#include <hip/hip_runtime.h>
#include <hip/hip_bf16.h>

// ---------------------------------------------------------------------------
// Types / helpers
// ---------------------------------------------------------------------------
typedef __attribute__((ext_vector_type(16))) __bf16 v16bf;
typedef __attribute__((ext_vector_type(8)))  float  v8f;

struct U4x2 { uint4 lo, hi; };

static __device__ __forceinline__ uint4 ld16u(const unsigned short* p) {
    return *reinterpret_cast<const uint4*>(p);
}
static __device__ __forceinline__ v16bf make_frag(uint4 lo, uint4 hi) {
    U4x2 t{lo, hi};
    return __builtin_bit_cast(v16bf, t);
}
static __device__ __forceinline__ unsigned short f2bf_u16(float f) {
    unsigned int u = __builtin_bit_cast(unsigned int, f);
    unsigned int r = u + 0x7FFFu + ((u >> 16) & 1u); // round-to-nearest-even
    return (unsigned short)(r >> 16);
}

// ---------------------------------------------------------------------------
// fp32 -> bf16 convert
// ---------------------------------------------------------------------------
__global__ void cvt_bf16_kernel(const float* __restrict__ in,
                                unsigned short* __restrict__ out, int n) {
    int i = blockIdx.x * 256 + threadIdx.x;
    if (i < n) out[i] = f2bf_u16(in[i]);
}

// ---------------------------------------------------------------------------
// GEMM: C[M,N] = A[M,K](bf16, row-major) * W[N,K](bf16, row-major)^T + bias
// block = 256 threads (8 waves). Tile: 128(M) x 128(N), K step 64.
// Wave w computes rows [w*16, w*16+16). B tile ping-pong staged in LDS
// (2 x 16KB) -> single barrier per K-step, copy of step k+1 overlaps the
// 16-WMMA chain of step k. global_prefetch pulls step k+2 toward L2.
// ---------------------------------------------------------------------------
__launch_bounds__(256)
__global__ void gemm_bf16_wmma(const unsigned short* __restrict__ A,
                               const unsigned short* __restrict__ W,
                               const float* __restrict__ bias,
                               float* __restrict__ C,
                               int M, int N, int K) {
    __shared__ alignas(16) unsigned short sB[2][128 * 64]; // [buf][n_local][k_local]

    const int tid  = threadIdx.x;
    const int wave = tid >> 5;
    const int lane = tid & 31;
    const int l16  = lane & 15;
    const int hi   = lane >> 4;           // 0: lanes 0-15, 1: lanes 16-31
    const int aoff = hi * 8;              // A-fragment K offset per ISA layout

    const int m0 = blockIdx.y * 128 + wave * 16;
    const int n0 = blockIdx.x * 128;

    v8f acc[8];
#pragma unroll
    for (int i = 0; i < 8; ++i) acc[i] = v8f{};

    const unsigned short* Arow = A + (size_t)(m0 + l16) * K;

    // stage 128 rows x 64 k of W into sB[buf]; 64B per thread
    const int sn   = tid >> 1;
    const int shal = tid & 1;
    auto stage = [&](int buf, int k0) {
        const uint4* src =
            reinterpret_cast<const uint4*>(W + (size_t)(n0 + sn) * K + k0 + shal * 32);
        uint4* dst = reinterpret_cast<uint4*>(&sB[buf][sn * 64 + shal * 32]);
        dst[0] = src[0]; dst[1] = src[1]; dst[2] = src[2]; dst[3] = src[3];
    };

    stage(0, 0);
    __syncthreads();

    const int nsteps = K >> 6; // K/64
    for (int step = 0; step < nsteps; ++step) {
        const int k0  = step * 64;
        const int buf = step & 1;

        if (step + 1 < nsteps) {
            stage(buf ^ 1, k0 + 64);
            if (k0 + 128 < K) { // pull step+2 toward L2 (global_prefetch_b8)
                __builtin_prefetch(W + (size_t)(n0 + sn) * K + k0 + 128, 0, 0);
                __builtin_prefetch(Arow + k0 + 128, 0, 0);
            }
        }

        // A fragments (16x32 each): elems[0:8]=K[aoff..), elems[8:16]=K[aoff+16..)
        v16bf a0 = make_frag(ld16u(Arow + k0 + aoff),
                             ld16u(Arow + k0 + aoff + 16));
        v16bf a1 = make_frag(ld16u(Arow + k0 + 32 + aoff),
                             ld16u(Arow + k0 + 32 + aoff + 16));

#pragma unroll
        for (int nt = 0; nt < 8; ++nt) {
            // B fragments (32x16): N = l16, elems[0:8]=K[hi*8..), elems[8:16]=K[hi*8+16..)
            const unsigned short* bp = &sB[buf][(nt * 16 + l16) * 64 + hi * 8];
            v16bf b0 = make_frag(ld16u(bp),      ld16u(bp + 16));
            v16bf b1 = make_frag(ld16u(bp + 32), ld16u(bp + 48));
            acc[nt] = __builtin_amdgcn_wmma_f32_16x16x32_bf16(
                false, a0, false, b0, (short)0, acc[nt], false, false);
            acc[nt] = __builtin_amdgcn_wmma_f32_16x16x32_bf16(
                false, a1, false, b1, (short)0, acc[nt], false, false);
        }
        __syncthreads(); // readers of sB[buf] done; sB[buf] is next staging target
    }

    // epilogue: C layout -> M = r + hi*8, N = nt*16 + l16
#pragma unroll
    for (int nt = 0; nt < 8; ++nt) {
        int n = n0 + nt * 16 + l16;
        float bv = bias[n];
#pragma unroll
        for (int r = 0; r < 8; ++r) {
            int m = m0 + r + hi * 8;
            C[(size_t)m * N + n] = acc[nt][r] + bv;
        }
    }
}

// ---------------------------------------------------------------------------
// RoPE + repack: qkv fp32 [B*S, 3072] ->
//   qr,kr bf16 [B*H, S, 64] (roped),  vT bf16 [B*H, 64, S]
// ---------------------------------------------------------------------------
__global__ void rope_pack_kernel(const float* __restrict__ qkv,
                                 unsigned short* __restrict__ qr,
                                 unsigned short* __restrict__ kr,
                                 unsigned short* __restrict__ vT) {
    const int S = 2048;
    int idx = blockIdx.x * 256 + threadIdx.x;
    int d  = idx & 63;
    int s  = (idx >> 6) & (S - 1);
    int bh = idx >> 17;                   // 0..31
    int b  = bh >> 4, h = bh & 15;

    const float* row = qkv + (size_t)(b * S + s) * 3072;
    const float* qh  = row + h * 64;
    const float* kh  = row + 1024 + h * 64;
    const float* vh  = row + 2048 + h * 64;

    float qo, ko;
    if (d < 32) {
        float ang = (float)s * __powf(10000.f, -(float)d * (1.f / 32.f));
        float sn = __sinf(ang), cs = __cosf(ang);
        qo = qh[d] * cs - qh[d + 32] * sn;
        ko = kh[d] * cs - kh[d + 32] * sn;
    } else {
        int dd = d - 32;
        float ang = (float)s * __powf(10000.f, -(float)dd * (1.f / 32.f));
        float sn = __sinf(ang), cs = __cosf(ang);
        qo = qh[dd] * sn + qh[d] * cs;
        ko = kh[dd] * sn + kh[d] * cs;
    }
    size_t qi = ((size_t)bh * S + s) * 64 + d;
    qr[qi] = f2bf_u16(qo);
    kr[qi] = f2bf_u16(ko);
    vT[((size_t)bh * 64 + d) * S + s] = f2bf_u16(vh[d]);
}

// ---------------------------------------------------------------------------
// Flash attention (causal): grid = (B*H, S/128), block = 256 (8 waves).
// Wave w owns 16 query rows. Per 32-key block: 4 score WMMAs, online softmax
// (16-lane shfl reductions), P staged bf16 in wave-private LDS, 4 PV WMMAs.
// Output bf16 [B*S, 1024].
// ---------------------------------------------------------------------------
__launch_bounds__(256)
__global__ void flash_attn_wmma(const unsigned short* __restrict__ qr,
                                const unsigned short* __restrict__ kr,
                                const unsigned short* __restrict__ vT,
                                unsigned short* __restrict__ ob) {
    const int S = 2048;
    __shared__ alignas(16) unsigned short sP[8][16 * 32]; // per-wave P tile

    const int tid  = threadIdx.x;
    const int wave = tid >> 5;
    const int lane = tid & 31;
    const int l16  = lane & 15;
    const int hi   = lane >> 4;
    const int aoff = hi * 8;

    const int bh = blockIdx.x;
    const int q0 = blockIdx.y * 128 + wave * 16;

    const unsigned short* Qb = qr + (size_t)bh * S * 64;
    const unsigned short* Kb = kr + (size_t)bh * S * 64;
    const unsigned short* Vb = vT + (size_t)bh * 64 * S;
    unsigned short* myP = &sP[wave][0];

    // Q fragments for d 0..31 and d 32..63 (A-matrix 16x32 layout)
    const unsigned short* qrow = Qb + (size_t)(q0 + l16) * 64;
    v16bf qa0 = make_frag(ld16u(qrow + aoff),      ld16u(qrow + aoff + 16));
    v16bf qa1 = make_frag(ld16u(qrow + 32 + aoff), ld16u(qrow + 32 + aoff + 16));

    v8f acc[4];
#pragma unroll
    for (int i = 0; i < 4; ++i) acc[i] = v8f{};
    float m_r[8], l_r[8];
#pragma unroll
    for (int r = 0; r < 8; ++r) { m_r[r] = -1e30f; l_r[r] = 0.f; }

    const float scale = 0.125f; // 1/sqrt(64)
    const int kbmax = (q0 + 15) >> 5; // last 32-key block touching this wave's rows

    for (int kb = 0; kb <= kbmax; ++kb) {
        const int kbase = kb * 32;

        // ---- scores: two 16x16 tiles over d = 0..63 ----
        v8f sc[2];
#pragma unroll
        for (int t = 0; t < 2; ++t) {
            int key = kbase + t * 16 + l16;
            const unsigned short* kp = Kb + (size_t)key * 64;
            v16bf bk0 = make_frag(ld16u(kp + hi * 8),      ld16u(kp + hi * 8 + 16));
            v16bf bk1 = make_frag(ld16u(kp + 32 + hi * 8), ld16u(kp + 32 + hi * 8 + 16));
            v8f c = v8f{};
            c = __builtin_amdgcn_wmma_f32_16x16x32_bf16(false, qa0, false, bk0,
                                                        (short)0, c, false, false);
            c = __builtin_amdgcn_wmma_f32_16x16x32_bf16(false, qa1, false, bk1,
                                                        (short)0, c, false, false);
            sc[t] = c;
        }

        // ---- online softmax update (per accumulator row r) ----
        int key0 = kbase + l16;
        int key1 = kbase + 16 + l16;
#pragma unroll
        for (int r = 0; r < 8; ++r) {
            int qrow_g = q0 + r + hi * 8;
            float v0 = (key0 > qrow_g) ? -1e9f : sc[0][r] * scale;
            float v1 = (key1 > qrow_g) ? -1e9f : sc[1][r] * scale;

            float mt = fmaxf(v0, v1);
            for (int off = 1; off < 16; off <<= 1)
                mt = fmaxf(mt, __shfl_xor(mt, off, 32));
            float mnew = fmaxf(m_r[r], mt);
            float corr = __expf(m_r[r] - mnew);
            float p0 = __expf(v0 - mnew);
            float p1 = __expf(v1 - mnew);
            float rs = p0 + p1;
            for (int off = 1; off < 16; off <<= 1)
                rs += __shfl_xor(rs, off, 32);
            l_r[r] = l_r[r] * corr + rs;
            m_r[r] = mnew;
            acc[0][r] *= corr; acc[1][r] *= corr;
            acc[2][r] *= corr; acc[3][r] *= corr;

            int rl = r + hi * 8;
            myP[rl * 32 + l16]      = f2bf_u16(p0);
            myP[rl * 32 + 16 + l16] = f2bf_u16(p1);
        }
        // wave-private LDS: order DS writes before cross-lane DS reads
        asm volatile("s_wait_dscnt 0" ::: "memory");

        // ---- PV: P(16x32 keys) x V(32 keys x 64 d) ----
        const unsigned short* pp = myP + l16 * 32;
        v16bf pa = make_frag(ld16u(pp + aoff), ld16u(pp + aoff + 16));
#pragma unroll
        for (int dt = 0; dt < 4; ++dt) {
            const unsigned short* vp =
                Vb + (size_t)(dt * 16 + l16) * S + kbase + hi * 8;
            v16bf vb = make_frag(ld16u(vp), ld16u(vp + 16));
            acc[dt] = __builtin_amdgcn_wmma_f32_16x16x32_bf16(
                false, pa, false, vb, (short)0, acc[dt], false, false);
        }
        // DS reads done before next iteration overwrites P
        asm volatile("s_wait_dscnt 0" ::: "memory");
    }

    // ---- epilogue: out bf16 [B*S, 1024] ----
    const int b = bh >> 4, h = bh & 15;
#pragma unroll
    for (int r = 0; r < 8; ++r) {
        int srow = q0 + r + hi * 8;
        float inv = 1.f / l_r[r];
#pragma unroll
        for (int dt = 0; dt < 4; ++dt) {
            int d = dt * 16 + l16;
            ob[(size_t)(b * S + srow) * 1024 + h * 64 + d] =
                f2bf_u16(acc[dt][r] * inv);
        }
    }
}

// ---------------------------------------------------------------------------
// Host orchestration
// ---------------------------------------------------------------------------
extern "C" void kernel_launch(void* const* d_in, const int* in_sizes, int n_in,
                              void* d_out, int out_size, void* d_ws, size_t ws_size,
                              hipStream_t stream) {
    const int B = 2, S = 2048, DIM = 1024, H = 16;
    const int M = B * S;            // 4096
    const int K = DIM;              // 1024
    const int N1 = 3 * DIM;         // 3072

    const float* x      = (const float*)d_in[0];
    const float* qkv_w  = (const float*)d_in[1];
    const float* qkv_b  = (const float*)d_in[2];
    const float* proj_w = (const float*)d_in[3];
    const float* proj_b = (const float*)d_in[4];
    float* out = (float*)d_out;

    char* ws = (char*)d_ws;
    size_t off = 0;
    auto carve = [&](size_t bytes) { char* p = ws + off; off += (bytes + 255) & ~(size_t)255; return p; };
    unsigned short* xb   = (unsigned short*)carve((size_t)M * K * 2);       // x bf16
    unsigned short* wqb  = (unsigned short*)carve((size_t)N1 * K * 2);      // qkv_w bf16
    unsigned short* wpb  = (unsigned short*)carve((size_t)DIM * K * 2);     // proj_w bf16
    float*          qkv  = (float*)carve((size_t)M * N1 * 4);               // qkv fp32
    unsigned short* q_r  = (unsigned short*)carve((size_t)B * H * S * 64 * 2);
    unsigned short* k_r  = (unsigned short*)carve((size_t)B * H * S * 64 * 2);
    unsigned short* v_t  = (unsigned short*)carve((size_t)B * H * 64 * S * 2);
    unsigned short* o_b  = (unsigned short*)carve((size_t)M * DIM * 2);

    // 1) converts
    {
        int n;
        n = M * K;   cvt_bf16_kernel<<<(n + 255) / 256, 256, 0, stream>>>(x, xb, n);
        n = N1 * K;  cvt_bf16_kernel<<<(n + 255) / 256, 256, 0, stream>>>(qkv_w, wqb, n);
        n = DIM * K; cvt_bf16_kernel<<<(n + 255) / 256, 256, 0, stream>>>(proj_w, wpb, n);
    }
    // 2) QKV GEMM: qkv = x @ qkv_w^T + qkv_b
    gemm_bf16_wmma<<<dim3(N1 / 128, M / 128), 256, 0, stream>>>(
        xb, wqb, qkv_b, qkv, M, N1, K);
    // 3) RoPE + repack
    {
        int n = B * H * S * 64; // 4M threads
        rope_pack_kernel<<<n / 256, 256, 0, stream>>>(qkv, q_r, k_r, v_t);
    }
    // 4) flash attention
    flash_attn_wmma<<<dim3(B * H, S / 128), 256, 0, stream>>>(q_r, k_r, v_t, o_b);
    // 5) projection GEMM: out = o @ proj_w^T + proj_b
    gemm_bf16_wmma<<<dim3(DIM / 128, M / 128), 256, 0, stream>>>(
        o_b, wpb, proj_b, out, M, DIM, K);
}
